// MoE_71579924955713
// MI455X (gfx1250) — matrix operand order repaired and checked by reading, not compiled, not run on hardware
//
#include <hip/hip_runtime.h>
#include <hip/hip_bf16.h>

// ---------------------------------------------------------------------------
// MoE (top-2 of 8 experts), D=1024, H=2048, N=4096 tokens, fp32 in/out.
// Pipeline:
//   0) fp32 -> bf16 pre-convert of x, W1, W2 (halves streaming bytes)
//   1) gating (wave32 per token)
//   2) deterministic per-expert dispatch lists (ballot compaction)
//   3) expert FFN: 32-token tiles, bf16 WMMA, X staged via
//      global_load_async_to_lds_b128 (ASYNCcnt engine), 192KB LDS/block,
//      4 column-tiles per accumulation pass -> 8 WMMAs per A-fragment load
//   4) deterministic combine, 5) aux losses
// ---------------------------------------------------------------------------

typedef __attribute__((ext_vector_type(16))) __bf16 v16bf;
typedef __attribute__((ext_vector_type(8)))  float  v8f;

#define MOE_E 8
#define MOE_D 1024
#define MOE_H 2048
#define TILE_M 32

// CDNA5 async global->LDS copy (16B per lane), tracked by ASYNCcnt.
__device__ __forceinline__ void async_copy16(unsigned lds_off, const void* g) {
    asm volatile("global_load_async_to_lds_b128 %0, %1, off"
                 :: "v"(lds_off), "v"((unsigned long long)(uintptr_t)g)
                 : "memory");
}
__device__ __forceinline__ void wait_async0() {
    asm volatile("s_wait_asynccnt 0x0" ::: "memory");
}

// branch-free GELU (sigmoid form): no exec-mask divergence in the hot loop
__device__ __forceinline__ float gelu_f(float v) {
    return v / (1.0f + __expf(-1.702f * v));
}

__device__ __forceinline__ v8f wmma_bf16(const v16bf& a, const v16bf& b,
                                         const v8f& c) {
    return __builtin_amdgcn_wmma_f32_16x16x32_bf16(
        false, a, false, b, (short)0, c, false, false);
}

// ---------------------------------------------------------------------------
// Kernel 0: fp32 -> bf16 convert (4 elements/thread, 8B packed stores).
// ---------------------------------------------------------------------------
struct bf4 { __bf16 v[4]; };
__global__ __launch_bounds__(256)
void moe_cvt_bf16(const float* __restrict__ src, __bf16* __restrict__ dst,
                  size_t n4)
{
    size_t i = (size_t)blockIdx.x * blockDim.x + threadIdx.x;
    const size_t stride = (size_t)gridDim.x * blockDim.x;
    for (; i < n4; i += stride) {
        const float4 f = *(const float4*)(src + i * 4);
        bf4 o;
        o.v[0] = (__bf16)f.x; o.v[1] = (__bf16)f.y;
        o.v[2] = (__bf16)f.z; o.v[3] = (__bf16)f.w;
        *(bf4*)(dst + i * 4) = o;
    }
}

// ---------------------------------------------------------------------------
// Kernel 1: gating. One wave32 per token. W_gate staged in LDS (32KB).
// ---------------------------------------------------------------------------
__global__ __launch_bounds__(256)
void moe_gate_kernel(const float* __restrict__ x,
                     const float* __restrict__ wg,
                     float* __restrict__ probs,
                     float* __restrict__ lse,
                     int* __restrict__ top_i,
                     float* __restrict__ top_v)
{
    __shared__ float wgs[MOE_E * MOE_D];     // 32 KB
    const int tid = threadIdx.x;
    for (int i = tid; i < MOE_E * MOE_D; i += 256) wgs[i] = wg[i];
    __syncthreads();

    const int wave = tid >> 5;
    const int lane = tid & 31;
    const int n = blockIdx.x * 8 + wave;

    float p[MOE_E];
#pragma unroll
    for (int e = 0; e < MOE_E; ++e) p[e] = 0.0f;

    const float* xr = x + (size_t)n * MOE_D;
    for (int d = lane; d < MOE_D; d += 32) {
        const float xv = xr[d];
#pragma unroll
        for (int e = 0; e < MOE_E; ++e) p[e] += xv * wgs[e * MOE_D + d];
    }
#pragma unroll
    for (int e = 0; e < MOE_E; ++e) {
        float v = p[e];
        for (int off = 16; off > 0; off >>= 1) v += __shfl_xor(v, off, 32);
        p[e] = v;
    }
    float mx = p[0];
#pragma unroll
    for (int e = 1; e < MOE_E; ++e) mx = fmaxf(mx, p[e]);
    float pr[MOE_E]; float s = 0.0f;
#pragma unroll
    for (int e = 0; e < MOE_E; ++e) { pr[e] = __expf(p[e] - mx); s += pr[e]; }
    const float inv = 1.0f / s;
#pragma unroll
    for (int e = 0; e < MOE_E; ++e) pr[e] *= inv;
    const float l = mx + __logf(s);
    int i0 = 0;
#pragma unroll
    for (int e = 1; e < MOE_E; ++e) if (pr[e] > pr[i0]) i0 = e;
    int i1 = (i0 == 0) ? 1 : 0;
#pragma unroll
    for (int e = 0; e < MOE_E; ++e) if (e != i0 && pr[e] > pr[i1]) i1 = e;

    if (lane < MOE_E) probs[(size_t)n * MOE_E + lane] = pr[lane];
    if (lane == 0) {
        lse[n] = l;
        top_i[n * 2 + 0] = i0;  top_i[n * 2 + 1] = i1;
        top_v[n * 2 + 0] = pr[i0]; top_v[n * 2 + 1] = pr[i1];
    }
}

// ---------------------------------------------------------------------------
// Kernel 2: deterministic per-expert dispatch lists (ballot compaction).
// ---------------------------------------------------------------------------
__global__ __launch_bounds__(256)
void moe_build_lists(const int* __restrict__ top_i,
                     const float* __restrict__ top_v,
                     int* __restrict__ cnt,
                     int* __restrict__ tokL,
                     int* __restrict__ slotL,
                     float* __restrict__ gateL,
                     int N)
{
    const int e    = threadIdx.x >> 5;   // 8 waves = 8 experts
    const int lane = threadIdx.x & 31;
    int base = 0;
    for (int n0 = 0; n0 < N; n0 += 32) {
        const int n = n0 + lane;
        const int a = top_i[n * 2 + 0];
        const int b = top_i[n * 2 + 1];
        const bool h0 = (a == e), h1 = (b == e);
        const bool hit = h0 || h1;
        const unsigned mask = (unsigned)__ballot(hit);
        const int pre = __popc(mask & ((1u << lane) - 1u));
        if (hit) {
            const int pos = base + pre;
            tokL [e * N + pos] = n;
            slotL[e * N + pos] = h0 ? 0 : 1;
            gateL[e * N + pos] = h0 ? top_v[n * 2 + 0] : top_v[n * 2 + 1];
        }
        base += __popc(mask);
    }
    if (lane == 0) cnt[e] = base;
}

// ---------------------------------------------------------------------------
// Kernel 3: expert FFN. grid = (N/32 tiles, 8 experts); 256 thr = 8 waves.
// 32 gathered tokens per block; each wave owns a column strip. Column tiles
// are processed 4-at-a-time so one A-fragment LDS load feeds 8 WMMAs.
// ---------------------------------------------------------------------------
__global__ __launch_bounds__(256)
void moe_expert_kernel(const __bf16* __restrict__ xb,   // [N][D] bf16
                       const __bf16* __restrict__ w1b,  // [E][D][H] bf16
                       const __bf16* __restrict__ w2b,  // [E][H][D] bf16
                       const int* __restrict__ cnt,
                       const int* __restrict__ tokL,
                       const int* __restrict__ slotL,
                       const float* __restrict__ gateL,
                       float* __restrict__ contrib,     // [2][N][D]
                       int N)
{
    __shared__ __align__(16) __bf16 xs[TILE_M * MOE_D]; //  64 KB
    __shared__ __align__(16) __bf16 hs[TILE_M * MOE_H]; // 128 KB
    __shared__ int    s_tok[TILE_M];
    __shared__ int    s_slot[TILE_M];
    __shared__ float  s_gate[TILE_M];
    __shared__ int    s_valid[TILE_M];

    const int e = blockIdx.y;
    const int t = blockIdx.x;
    const int c = cnt[e];
    if (t * TILE_M >= c) return;            // uniform per block

    const int tid  = threadIdx.x;
    const int lane = tid & 31;
    const int wave = tid >> 5;

    if (tid < TILE_M) {
        const int idx = t * TILE_M + tid;
        const bool ok = idx < c;
        s_valid[tid] = ok ? 1 : 0;
        s_tok[tid]   = ok ? tokL [e * N + idx] : 0;
        s_slot[tid]  = ok ? slotL[e * N + idx] : 0;
        s_gate[tid]  = ok ? gateL[e * N + idx] : 0.0f;
    }
    __syncthreads();

    // ---- stage A: async-DMA gather of 32 bf16 token rows into LDS ---------
    {
        const unsigned xs_base = (unsigned)(uintptr_t)&xs[0];
        for (int tr = tid; tr < TILE_M * 128; tr += 256) {
            const int row  = tr >> 7;            // 128 x 16B per row
            const int boff = (tr & 127) * 16;    // byte offset in row
            const char* g = (const char*)(xb + (size_t)s_tok[row] * MOE_D) + boff;
            async_copy16(xs_base + row * (MOE_D * 2) + boff, g);
        }
        wait_async0();
    }
    __syncthreads();

    const int m_a = lane & 15;               // A-frag: M = lane % 16
    const int kb  = (lane < 16) ? 0 : 8;     // A-frag K-phase (ISA layout)
    const int n_c = lane & 15;               // C tile: N = lane % 16
    const int rb  = (lane < 16) ? 0 : 8;     // C tile: row base
    const v8f vzero = {};

    // ---- GEMM1: h[32][2048] = gelu(X @ W1[e]); wave owns 256 H-cols -------
    const __bf16* w1e = w1b + (size_t)e * MOE_D * MOE_H;
    for (int hg = 0; hg < 4; ++hg) {
        const int colg = wave * 256 + hg * 64;       // 4 tiles of 16 columns
        v8f acc0[4], acc1[4];
#pragma unroll
        for (int j = 0; j < 4; ++j) { acc0[j] = vzero; acc1[j] = vzero; }
        for (int k0 = 0; k0 < MOE_D; k0 += 32) {
            // A fragments once per k-step (4x ds_load_b128), feed 8 WMMAs
            v16bf a0, a1;
            const __bf16* p0 = &xs[m_a * MOE_D + k0];
            const __bf16* p1 = &xs[(m_a + 16) * MOE_D + k0];
            *(uint4*)&a0       = *(const uint4*)(p0 + kb);
            *((uint4*)&a0 + 1) = *(const uint4*)(p0 + 16 + kb);
            *(uint4*)&a1       = *(const uint4*)(p1 + kb);
            *((uint4*)&a1 + 1) = *(const uint4*)(p1 + 16 + kb);
            // B: lane owns column lane%16; K in same table order as A
            const __bf16* brow = w1e + (size_t)k0 * MOE_H + colg + n_c;
            __builtin_prefetch(brow + 32 * MOE_H, 0, 1);
#pragma unroll
            for (int j = 0; j < 4; ++j) {
                v16bf b;
#pragma unroll
                for (int s = 0; s < 16; ++s) b[s] = brow[(size_t)(kb + (s & 7) + ((s >> 3) << 4)) * MOE_H + j * 16];
                acc0[j] = wmma_bf16(a0, b, acc0[j]);
                acc1[j] = wmma_bf16(a1, b, acc1[j]);
            }
        }
#pragma unroll
        for (int j = 0; j < 4; ++j) {
            const int col0 = colg + j * 16;
#pragma unroll
            for (int r = 0; r < 8; ++r) {
                hs[(r + rb)      * MOE_H + col0 + n_c] = (__bf16)gelu_f(acc0[j][r]);
                hs[(r + rb + 16) * MOE_H + col0 + n_c] = (__bf16)gelu_f(acc1[j][r]);
            }
        }
    }
    __syncthreads();

    // ---- GEMM2: y[32][1024] = h @ W2[e]; wave owns 128 D-cols -------------
    const __bf16* w2e = w2b + (size_t)e * MOE_H * MOE_D;
    for (int dg = 0; dg < 2; ++dg) {
        const int colg = wave * 128 + dg * 64;
        v8f acc0[4], acc1[4];
#pragma unroll
        for (int j = 0; j < 4; ++j) { acc0[j] = vzero; acc1[j] = vzero; }
        for (int k0 = 0; k0 < MOE_H; k0 += 32) {
            v16bf a0, a1;
            const __bf16* p0 = &hs[m_a * MOE_H + k0];
            const __bf16* p1 = &hs[(m_a + 16) * MOE_H + k0];
            *(uint4*)&a0       = *(const uint4*)(p0 + kb);
            *((uint4*)&a0 + 1) = *(const uint4*)(p0 + 16 + kb);
            *(uint4*)&a1       = *(const uint4*)(p1 + kb);
            *((uint4*)&a1 + 1) = *(const uint4*)(p1 + 16 + kb);
            const __bf16* brow = w2e + (size_t)k0 * MOE_D + colg + n_c;
            __builtin_prefetch(brow + 32 * MOE_D, 0, 1);
#pragma unroll
            for (int j = 0; j < 4; ++j) {
                v16bf b;
#pragma unroll
                for (int s = 0; s < 16; ++s) b[s] = brow[(size_t)(kb + (s & 7) + ((s >> 3) << 4)) * MOE_D + j * 16];
                acc0[j] = wmma_bf16(a0, b, acc0[j]);
                acc1[j] = wmma_bf16(a1, b, acc1[j]);
            }
        }
#pragma unroll
        for (int j = 0; j < 4; ++j) {
            const int col0 = colg + j * 16;
#pragma unroll
            for (int r = 0; r < 8; ++r) {
                const int m0 = r + rb;
                const int m1 = r + rb + 16;
                if (s_valid[m0])
                    contrib[((size_t)s_slot[m0] * N + s_tok[m0]) * MOE_D + col0 + n_c]
                        = acc0[j][r] * s_gate[m0];
                if (s_valid[m1])
                    contrib[((size_t)s_slot[m1] * N + s_tok[m1]) * MOE_D + col0 + n_c]
                        = acc1[j][r] * s_gate[m1];
            }
        }
    }
}

// ---------------------------------------------------------------------------
// Kernel 4: y = contrib0 + contrib1 (deterministic combine, no atomics).
// ---------------------------------------------------------------------------
__global__ __launch_bounds__(256)
void moe_combine(const float* __restrict__ c0,
                 const float* __restrict__ c1,
                 float* __restrict__ y, size_t total)
{
    size_t i = (size_t)blockIdx.x * blockDim.x + threadIdx.x;
    const size_t stride = (size_t)gridDim.x * blockDim.x;
    for (; i < total; i += stride) y[i] = c0[i] + c1[i];
}

// ---------------------------------------------------------------------------
// Kernel 5: aux losses, fixed-order tree reduction (deterministic).
// ---------------------------------------------------------------------------
__global__ __launch_bounds__(256)
void moe_loss_kernel(const float* __restrict__ probs,
                     const float* __restrict__ lse,
                     const int* __restrict__ top_i,
                     const float* __restrict__ top_v,
                     float* __restrict__ out_loss, int N)
{
    __shared__ float red[256];
    const int tid = threadIdx.x;
    float pc[MOE_E], gc[MOE_E], fc[MOE_E], z = 0.0f;
#pragma unroll
    for (int e = 0; e < MOE_E; ++e) { pc[e] = 0.f; gc[e] = 0.f; fc[e] = 0.f; }

    for (int n = tid; n < N; n += 256) {
#pragma unroll
        for (int e = 0; e < MOE_E; ++e) pc[e] += probs[(size_t)n * MOE_E + e];
        const int i0 = top_i[n * 2 + 0], i1 = top_i[n * 2 + 1];
        const float v0 = top_v[n * 2 + 0], v1 = top_v[n * 2 + 1];
#pragma unroll
        for (int e = 0; e < MOE_E; ++e) {
            gc[e] += (i0 == e ? v0 : 0.f) + (i1 == e ? v1 : 0.f);
            fc[e] += (i0 == e ? 1.f : 0.f) + (i1 == e ? 1.f : 0.f);
        }
        const float l = lse[n];
        z += l * l;
    }

    float PC[MOE_E], GC[MOE_E], FC[MOE_E], Z;
#pragma unroll
    for (int e = 0; e < MOE_E; ++e) {
        red[tid] = pc[e]; __syncthreads();
        for (int s = 128; s > 0; s >>= 1) { if (tid < s) red[tid] += red[tid + s]; __syncthreads(); }
        PC[e] = red[0]; __syncthreads();
        red[tid] = gc[e]; __syncthreads();
        for (int s = 128; s > 0; s >>= 1) { if (tid < s) red[tid] += red[tid + s]; __syncthreads(); }
        GC[e] = red[0]; __syncthreads();
        red[tid] = fc[e]; __syncthreads();
        for (int s = 128; s > 0; s >>= 1) { if (tid < s) red[tid] += red[tid + s]; __syncthreads(); }
        FC[e] = red[0]; __syncthreads();
    }
    red[tid] = z; __syncthreads();
    for (int s = 128; s > 0; s >>= 1) { if (tid < s) red[tid] += red[tid + s]; __syncthreads(); }
    Z = red[0];

    if (tid == 0) {
        float s1 = 0.f;
        for (int e = 0; e < MOE_E; ++e) s1 += fabsf(GC[e]);
        s1 = fmaxf(s1, 1e-12f);
        float mean = 0.f;
        float g[MOE_E];
        for (int e = 0; e < MOE_E; ++e) { g[e] = GC[e] / s1; mean += g[e]; }
        mean *= (1.0f / MOE_E);
        float var = 0.f;
        for (int e = 0; e < MOE_E; ++e) { const float d = g[e] - mean; var += d * d; }
        var *= (1.0f / MOE_E);
        const float cv = var / (mean * mean + 1e-10f);
        float sp = 0.f, sf = 0.f;
        for (int e = 0; e < MOE_E; ++e) { sp += fabsf(PC[e]); sf += fabsf(FC[e]); }
        sp = fmaxf(sp, 1e-12f); sf = fmaxf(sf, 1e-12f);
        float sw = 0.f;
        for (int e = 0; e < MOE_E; ++e) sw += (PC[e] / sp) * (FC[e] / sf);
        sw *= (float)MOE_E;
        const float zl = Z / (float)N;
        *out_loss = 0.01f * cv + 0.01f * sw + 0.001f * zl;
    }
}

// ---------------------------------------------------------------------------
// Launch. Workspace: bf16 copies of x/W1/W2 (75MB) + gating data + lists +
// contrib[2][N][D] (33.5MB).
// ---------------------------------------------------------------------------
extern "C" void kernel_launch(void* const* d_in, const int* in_sizes, int n_in,
                              void* d_out, int out_size, void* d_ws, size_t ws_size,
                              hipStream_t stream) {
    (void)n_in; (void)out_size; (void)ws_size;
    const float* x  = (const float*)d_in[0];
    const float* wg = (const float*)d_in[1];
    const float* w1 = (const float*)d_in[2];
    const float* w2 = (const float*)d_in[3];
    const int N = in_sizes[0] / MOE_D;      // 4096

    char* ws = (char*)d_ws;
    size_t off = 0;
    auto take = [&](size_t bytes) -> void* {
        void* p = (void*)(ws + off);
        off += (bytes + 255) & ~(size_t)255;
        return p;
    };
    const size_t nW = (size_t)MOE_E * MOE_D * MOE_H;
    __bf16* xbf  = (__bf16*)take((size_t)N * MOE_D * sizeof(__bf16));
    __bf16* w1b  = (__bf16*)take(nW * sizeof(__bf16));
    __bf16* w2b  = (__bf16*)take(nW * sizeof(__bf16));
    float* probs   = (float*)take((size_t)N * MOE_E * sizeof(float));
    float* lse     = (float*)take((size_t)N * sizeof(float));
    int*   top_i   = (int*)  take((size_t)N * 2 * sizeof(int));
    float* top_v   = (float*)take((size_t)N * 2 * sizeof(float));
    int*   cnt     = (int*)  take((size_t)MOE_E * sizeof(int));
    int*   tokL    = (int*)  take((size_t)MOE_E * N * sizeof(int));
    int*   slotL   = (int*)  take((size_t)MOE_E * N * sizeof(int));
    float* gateL   = (float*)take((size_t)MOE_E * N * sizeof(float));
    float* contrib = (float*)take((size_t)2 * N * MOE_D * sizeof(float));

    float* y    = (float*)d_out;
    float* loss = y + (size_t)N * MOE_D;

    moe_cvt_bf16<<<2048, 256, 0, stream>>>(x,  xbf, (size_t)N * MOE_D / 4);
    moe_cvt_bf16<<<4096, 256, 0, stream>>>(w1, w1b, nW / 4);
    moe_cvt_bf16<<<4096, 256, 0, stream>>>(w2, w2b, nW / 4);

    moe_gate_kernel<<<N / 8, 256, 0, stream>>>(x, wg, probs, lse, top_i, top_v);
    moe_build_lists<<<1, 256, 0, stream>>>(top_i, top_v, cnt, tokL, slotL, gateL, N);
    dim3 grid_e(N / TILE_M, MOE_E);
    moe_expert_kernel<<<grid_e, 256, 0, stream>>>(xbf, w1b, w2b, cnt, tokL, slotL,
                                                  gateL, contrib, N);
    moe_combine<<<1024, 256, 0, stream>>>(contrib, contrib + (size_t)N * MOE_D, y,
                                          (size_t)N * MOE_D);
    moe_loss_kernel<<<1, 256, 0, stream>>>(probs, lse, top_i, top_v, loss, N);
}
